// LSTMDecoder_53953379172846
// MI455X (gfx1250) — matrix operand order repaired and checked by reading, not compiled, not run on hardware
//
#include <hip/hip_runtime.h>
#include <hip/hip_bf16.h>
#include <math.h>

// Problem dims (from reference)
#define VOCAB 32000
#define EMB   512
#define HID   512
#define NLAY  2
#define BATCH 32
#define SEQ   64

typedef _Float16 v8h  __attribute__((ext_vector_type(8)));
typedef _Float16 v16h __attribute__((ext_vector_type(16)));
typedef float    v8f  __attribute__((ext_vector_type(8)));

// ---------------------------------------------------------------------------
// WMMA operand loaders (wave32), contiguous k order shared by A and B (not the ISA table order)
// A (16x32 f16, row-major src, leading dim ld):
//   lanes 0-15 : row = row0+lane,   elems = K {kb+0..15}
//   lanes 16-31: row = row0+lane-16, elems = K {kb+16..31}
// B (32x16 f16) built from W (N x K row-major, we need B[k][n] = W[n][k]):
//   lanes 0-15 : col = n0+lane,    elems = K {kb+0..15}
//   lanes 16-31: col = n0+lane-16, elems = K {kb+16..31}
// ---------------------------------------------------------------------------
__device__ __forceinline__ v16h ldA(const _Float16* __restrict__ base, int ld,
                                    int row0, int kb, int lane) {
  int r  = row0 + (lane & 15);
  int ks = kb + ((lane & 16) ? 16 : 0);
  const _Float16* p = base + r * ld + ks;
  v8h lo = *(const v8h*)p;
  v8h hi = *(const v8h*)(p + 8);
  return __builtin_shufflevector(lo, hi, 0,1,2,3,4,5,6,7,8,9,10,11,12,13,14,15);
}

__device__ __forceinline__ v16h ldB(const _Float16* __restrict__ W, int ld,
                                    int n0, int kb, int lane) {
  int n  = n0 + (lane & 15);
  int ks = kb + ((lane & 16) ? 16 : 0);
  const _Float16* p = W + n * ld + ks;
  v8h lo = *(const v8h*)p;
  v8h hi = *(const v8h*)(p + 8);
  return __builtin_shufflevector(lo, hi, 0,1,2,3,4,5,6,7,8,9,10,11,12,13,14,15);
}

__device__ __forceinline__ float sigf(float x) { return 1.0f / (1.0f + __expf(-x)); }

// ---------------------------------------------------------------------------
// f32 -> f16 conversion (one-time weight downconvert; streams through L2)
// ---------------------------------------------------------------------------
__global__ void __launch_bounds__(256) conv_f32_to_f16(const float* __restrict__ in,
                                                       _Float16* __restrict__ out, int n) {
  int i = blockIdx.x * blockDim.x + threadIdx.x;
  if (i < n) out[i] = (_Float16)in[i];
}

// x0 = input_seq[:, 0, :]  (strided gather + downconvert)
__global__ void __launch_bounds__(256) gather_x0_kernel(const float* __restrict__ inseq,
                                                        _Float16* __restrict__ x) {
  int i = blockIdx.x * blockDim.x + threadIdx.x;  // 0 .. BATCH*EMB-1
  if (i < BATCH * EMB) {
    int b = i >> 9;           // /EMB
    int e = i & (EMB - 1);
    x[i] = (_Float16)inseq[(size_t)b * SEQ * EMB + e];
  }
}

// ---------------------------------------------------------------------------
// Fused LSTM cell: one wave per 16 hidden units, computes all 4 gate tiles
// (32x16 each) with v_wmma_f32_16x16x32_f16, then the elementwise update.
// g = x @ Wih^T + bih + h @ Whh^T + bhh ; c' = sig(f)*c + sig(i)*tanh(g)
// h' = sig(o)*tanh(c')
// ---------------------------------------------------------------------------
__global__ void __launch_bounds__(32)
lstm_cell_kernel(const _Float16* __restrict__ x,    // BATCH x EMB (f16)
                 const _Float16* __restrict__ hin,  // BATCH x HID (f16)
                 _Float16* __restrict__ hout,       // BATCH x HID (f16)
                 float* __restrict__ c,             // BATCH x HID (f32)
                 const _Float16* __restrict__ Wih,  // 4H x EMB (f16)
                 const _Float16* __restrict__ Whh,  // 4H x HID (f16)
                 const float* __restrict__ bih,     // 4H
                 const float* __restrict__ bhh)     // 4H
{
  const int lane = threadIdx.x;
  const int j = blockIdx.x;                 // hidden-unit tile: columns j*16..j*16+15
  v8f acc[4][2] = {};                       // [gate][m-tile] 32x16 f32 accumulators

  // input contribution: x (32xEMB) @ Wih^T
  for (int kc = 0; kc < EMB; kc += 32) {
    v16h a0 = ldA(x, EMB, 0, kc, lane);
    v16h a1 = ldA(x, EMB, 16, kc, lane);
#pragma unroll
    for (int g = 0; g < 4; ++g) {
      v16h b = ldB(Wih, EMB, g * HID + j * 16, kc, lane);
      acc[g][0] = __builtin_amdgcn_wmma_f32_16x16x32_f16(false, a0, false, b,
                                                         (short)0, acc[g][0], false, false);
      acc[g][1] = __builtin_amdgcn_wmma_f32_16x16x32_f16(false, a1, false, b,
                                                         (short)0, acc[g][1], false, false);
    }
  }
  // recurrent contribution: h (32xHID) @ Whh^T
  for (int kc = 0; kc < HID; kc += 32) {
    v16h a0 = ldA(hin, HID, 0, kc, lane);
    v16h a1 = ldA(hin, HID, 16, kc, lane);
#pragma unroll
    for (int g = 0; g < 4; ++g) {
      v16h b = ldB(Whh, HID, g * HID + j * 16, kc, lane);
      acc[g][0] = __builtin_amdgcn_wmma_f32_16x16x32_f16(false, a0, false, b,
                                                         (short)0, acc[g][0], false, false);
      acc[g][1] = __builtin_amdgcn_wmma_f32_16x16x32_f16(false, a1, false, b,
                                                         (short)0, acc[g][1], false, false);
    }
  }

  // elementwise LSTM update — C/D layout: VGPR r holds M=r (lanes 0-15) and
  // M=r+8 (lanes 16-31); N = lane%16.
  const int hidx = j * 16 + (lane & 15);
  float bsum[4];
#pragma unroll
  for (int g = 0; g < 4; ++g) bsum[g] = bih[g * HID + hidx] + bhh[g * HID + hidx];
  const int mo = (lane & 16) ? 8 : 0;
#pragma unroll
  for (int mt = 0; mt < 2; ++mt) {
#pragma unroll
    for (int r = 0; r < 8; ++r) {
      int m = mt * 16 + r + mo;
      float gi = acc[0][mt][r] + bsum[0];
      float gf = acc[1][mt][r] + bsum[1];
      float gg = acc[2][mt][r] + bsum[2];
      float go = acc[3][mt][r] + bsum[3];
      float cold = c[m * HID + hidx];
      float cn   = sigf(gf) * cold + sigf(gi) * tanhf(gg);
      c[m * HID + hidx]    = cn;
      hout[m * HID + hidx] = (_Float16)(sigf(go) * tanhf(cn));
    }
  }
}

// ---------------------------------------------------------------------------
// FC head: out[:, t, :] = h @ fcW^T + fcb  (M=32, N=32000, K=512)
// 8 waves/block, one 32x16 output tile per wave. fcW streams from L2 in f16.
// ---------------------------------------------------------------------------
__global__ void __launch_bounds__(256)
fc_kernel(const _Float16* __restrict__ h,    // BATCH x HID (f16)
          const _Float16* __restrict__ fcW,  // VOCAB x HID (f16)
          const float* __restrict__ fcb,     // VOCAB
          float* __restrict__ out,           // BATCH x SEQ x VOCAB
          int t)
{
  const int lane  = threadIdx.x & 31;
  const int wave  = threadIdx.x >> 5;
  const int ntile = blockIdx.x * 8 + wave;   // 0..1999
  const int n0 = ntile * 16;
  v8f c0 = {}, c1 = {};

  const _Float16* bp = fcW + (size_t)(n0 + (lane & 15)) * HID + ((lane & 16) ? 16 : 0);
  for (int kc = 0; kc < HID; kc += 32) {
    __builtin_prefetch((const void*)(bp + kc + 128), 0, 0);   // global_prefetch next chunks
    v16h a0 = ldA(h, HID, 0, kc, lane);
    v16h a1 = ldA(h, HID, 16, kc, lane);
    v8h blo = *(const v8h*)(bp + kc);
    v8h bhi = *(const v8h*)(bp + kc + 8);
    v16h b = __builtin_shufflevector(blo, bhi, 0,1,2,3,4,5,6,7,8,9,10,11,12,13,14,15);
    c0 = __builtin_amdgcn_wmma_f32_16x16x32_f16(false, a0, false, b, (short)0, c0, false, false);
    c1 = __builtin_amdgcn_wmma_f32_16x16x32_f16(false, a1, false, b, (short)0, c1, false, false);
  }

  const int n = n0 + (lane & 15);
  const float bias = fcb[n];
  const int mo = (lane & 16) ? 8 : 0;
  float* o = out + (size_t)t * VOCAB;
#pragma unroll
  for (int r = 0; r < 8; ++r) {
    o[(size_t)(r + mo) * (SEQ * VOCAB) + n]      = c0[r] + bias;
    o[(size_t)(16 + r + mo) * (SEQ * VOCAB) + n] = c1[r] + bias;
  }
}

// ---------------------------------------------------------------------------
// Per-row argmax over VOCAB + embedding gather into next-step x (f16).
// One block per batch row; first-index tie-break to match jnp.argmax.
// ---------------------------------------------------------------------------
__global__ void __launch_bounds__(256)
argmax_embed_kernel(const float* __restrict__ out, const float* __restrict__ emb,
                    _Float16* __restrict__ x, int t)
{
  const int b = blockIdx.x;
  const int tid = threadIdx.x;
  const float* row = out + (size_t)b * SEQ * VOCAB + (size_t)t * VOCAB;

  float best = -INFINITY;
  int bidx = 0x7fffffff;
  for (int v = tid; v < VOCAB; v += 256) {
    float val = row[v];
    if (val > best) { best = val; bidx = v; }
  }

  __shared__ float sval[256];
  __shared__ int   sidx[256];
  sval[tid] = best; sidx[tid] = bidx;
  __syncthreads();
  for (int s = 128; s > 0; s >>= 1) {
    if (tid < s) {
      float v2 = sval[tid + s]; int i2 = sidx[tid + s];
      if (v2 > sval[tid] || (v2 == sval[tid] && i2 < sidx[tid])) {
        sval[tid] = v2; sidx[tid] = i2;
      }
    }
    __syncthreads();
  }
  const int idx = sidx[0];
  const float* e = emb + (size_t)idx * EMB;
  for (int k = tid; k < EMB; k += 256) x[b * EMB + k] = (_Float16)e[k];
}

// ---------------------------------------------------------------------------
// Host orchestration (graph-capture safe: only kernel launches + async d2d)
// ---------------------------------------------------------------------------
extern "C" void kernel_launch(void* const* d_in, const int* in_sizes, int n_in,
                              void* d_out, int out_size, void* d_ws, size_t ws_size,
                              hipStream_t stream) {
  const float* input_seq = (const float*)d_in[0];
  const float* h_encode  = (const float*)d_in[1];
  const float* c_encode  = (const float*)d_in[2];
  const float* Wih       = (const float*)d_in[3];
  const float* Whh       = (const float*)d_in[4];
  const float* bih       = (const float*)d_in[5];
  const float* bhh       = (const float*)d_in[6];
  const float* fcW       = (const float*)d_in[7];
  const float* fcb       = (const float*)d_in[8];
  const float* emb       = (const float*)d_in[9];
  float* out = (float*)d_out;

  // Workspace layout (all offsets 256B-aligned)
  char* ws = (char*)d_ws;
  size_t off = 0;
  _Float16* fcW_h = (_Float16*)(ws + off); off += (size_t)VOCAB * HID * 2;        // 32.77 MB
  _Float16* Wih_h = (_Float16*)(ws + off); off += (size_t)NLAY * 4 * HID * EMB * 2; // 4 MB
  _Float16* Whh_h = (_Float16*)(ws + off); off += (size_t)NLAY * 4 * HID * HID * 2; // 4 MB
  _Float16* hA    = (_Float16*)(ws + off); off += (size_t)NLAY * BATCH * HID * 2;
  _Float16* hB    = (_Float16*)(ws + off); off += (size_t)NLAY * BATCH * HID * 2;
  float*    c_ws  = (float*)(ws + off);    off += (size_t)NLAY * BATCH * HID * 4;
  _Float16* x_h   = (_Float16*)(ws + off); off += (size_t)BATCH * EMB * 2;

  // One-time downconversion + state init (re-done every call: deterministic)
  {
    int n;
    n = VOCAB * HID;
    conv_f32_to_f16<<<(n + 255) / 256, 256, 0, stream>>>(fcW, fcW_h, n);
    n = NLAY * 4 * HID * EMB;
    conv_f32_to_f16<<<(n + 255) / 256, 256, 0, stream>>>(Wih, Wih_h, n);
    n = NLAY * 4 * HID * HID;
    conv_f32_to_f16<<<(n + 255) / 256, 256, 0, stream>>>(Whh, Whh_h, n);
    n = NLAY * BATCH * HID;
    conv_f32_to_f16<<<(n + 255) / 256, 256, 0, stream>>>(h_encode, hA, n);
    hipMemcpyAsync(c_ws, c_encode, (size_t)NLAY * BATCH * HID * sizeof(float),
                   hipMemcpyDeviceToDevice, stream);
    n = BATCH * EMB;
    gather_x0_kernel<<<(n + 255) / 256, 256, 0, stream>>>(input_seq, x_h);
  }

  const int HL = BATCH * HID;  // per-layer h/c stride
  for (int t = 0; t < SEQ; ++t) {
    _Float16* hin  = (t & 1) ? hB : hA;   // double-buffered recurrent state
    _Float16* hout = (t & 1) ? hA : hB;

    // layer 0
    lstm_cell_kernel<<<HID / 16, 32, 0, stream>>>(
        x_h, hin, hout, c_ws, Wih_h, Whh_h, bih, bhh);
    // layer 1 (input = layer-0 new h)
    lstm_cell_kernel<<<HID / 16, 32, 0, stream>>>(
        hout, hin + HL, hout + HL, c_ws + HL,
        Wih_h + (size_t)4 * HID * EMB, Whh_h + (size_t)4 * HID * HID,
        bih + 4 * HID, bhh + 4 * HID);
    // logits for this step
    fc_kernel<<<(VOCAB / 16) / 8, 256, 0, stream>>>(hout + HL, fcW_h, fcb, out, t);
    // next token embedding -> x
    argmax_embed_kernel<<<BATCH, 256, 0, stream>>>(out, emb, x_h, t);
  }
}
